// MessageFunction_8031588844018
// MI455X (gfx1250) — matrix operand order repaired and checked by reading, not compiled, not run on hardware
//
#include <hip/hip_runtime.h>

typedef unsigned short u16;
typedef unsigned int   u32;
typedef float  v4f  __attribute__((ext_vector_type(4)));
typedef float  v8f  __attribute__((ext_vector_type(8)));
typedef u16    v8us __attribute__((ext_vector_type(8)));
typedef u16    v16us __attribute__((ext_vector_type(16)));
typedef __bf16 v16bf __attribute__((ext_vector_type(16)));

// Problem constants: B=1024, K=256 (E|NODE concat), M=128, N=512
#define NB       512
#define KB       256
#define MB       128
#define NTILE    128        // N columns per workgroup
#define KCHUNK   32         // one bf16-WMMA K-block per LDS stage
#define KSTRIDE  40         // [n][k] LDS plane: 32 + 8 pad elems; rows 80B (16B aligned)

// res = [hi16(a_hi), hi16(a_lo)]  (one v_perm_b32)
__device__ __forceinline__ u32 pack_hi16(u32 a_hi, u32 a_lo) {
    return __builtin_amdgcn_perm(a_hi, a_lo, 0x07060302u);
}

__device__ __forceinline__ v16bf join16(v8us lo, v8us hi) {
    v16us w = __builtin_shufflevector(lo, hi, 0,1,2,3,4,5,6,7,8,9,10,11,12,13,14,15);
    return __builtin_bit_cast(v16bf, w);
}

// ---- one-time prep: split W = [W_e | W_h] (f32) into bf16 hi/lo planes, [m][k] ----
__global__ __launch_bounds__(256)
void split_weights(const float* __restrict__ W_e, const float* __restrict__ W_h,
                   u16* __restrict__ Wh, u16* __restrict__ Wl)
{
    const int idx = blockIdx.x * 256 + threadIdx.x;    // 0..32767 = m*256 + k
    const int m = idx >> 8, k = idx & 255;
    const float x = (k < 128) ? W_e[m * 128 + k] : W_h[m * 128 + (k - 128)];
    const u32 u = __float_as_uint(x);
    const float hf = __uint_as_float(u & 0xffff0000u);
    Wh[idx] = (u16)(u >> 16);
    Wl[idx] = (u16)(__float_as_uint(x - hf) >> 16);
}

__global__ __launch_bounds__(256)
void msg_gemm_bf16x3(const float* __restrict__ h_w,
                     const float* __restrict__ e_vw,
                     const u16* __restrict__ Wh,
                     const u16* __restrict__ Wl,
                     const float* __restrict__ b_e,
                     const float* __restrict__ b_h,
                     float* __restrict__ out)
{
    __shared__ u16 Xh[NTILE * KSTRIDE];   // 10240 B bf16-hi plane, [n][k]
    __shared__ u16 Xl[NTILE * KSTRIDE];   // 10240 B bf16-lo plane

    const int n0 = blockIdx.x * NTILE;    // 0,128,256,384
    const int b  = blockIdx.y;            // 0..1023

    const int tid     = threadIdx.x;      // 8 waves of 32
    const int wave    = tid >> 5;
    const int lane    = tid & 31;
    const int lane_lo = lane & 15;
    const int hi      = lane >> 4;

    const int wave_m = wave >> 1;         // 0..3
    const int wave_n = wave & 1;          // 0..1
    const int m0     = wave_m * 32;       // two strips: m0, m0+16
    const int nw     = wave_n * 64;

    // staging roles: lane covers column sn, wave-half covers k-pair group
    const int sn = 32 * (wave & 3) + lane;  // 0..127
    const int kg = wave >> 2;               // 0..1

    // ---- accumulators (bias folded), C/D layout: VGPR i -> row base+i+8*hi ----
    v8f acc[2][4];
#pragma unroll
    for (int s = 0; s < 2; ++s)
#pragma unroll
        for (int i = 0; i < 8; ++i) {
            const int m   = m0 + s * 16 + i + 8 * hi;
            const float z = b_e[m] + b_h[m];
            acc[s][0][i] = z; acc[s][1][i] = z; acc[s][2][i] = z; acc[s][3][i] = z;
        }

    const size_t xbase = (size_t)b * (size_t)(128 * NB);

#pragma unroll 1
    for (int kc = 0; kc < KB / KCHUNK; ++kc) {
        const int k0 = kc * KCHUNK;
        const float* xsrc = (k0 < 128) ? (e_vw + xbase + (size_t)k0 * NB)
                                       : (h_w  + xbase + (size_t)(k0 - 128) * NB);

        __syncthreads();
        // ---- stage + split 32x128 slab: per thread 8 k-pairs at column sn ----
#pragma unroll
        for (int j = 0; j < 8; ++j) {
            const int k = (kg * 8 + j) * 2;                 // local k (even)
            const float x0 = xsrc[(size_t)k * NB + n0 + sn];
            const float x1 = xsrc[(size_t)(k + 1) * NB + n0 + sn];
            const u32 u0 = __float_as_uint(x0), u1 = __float_as_uint(x1);
            const float hf0 = __uint_as_float(u0 & 0xffff0000u);
            const float hf1 = __uint_as_float(u1 & 0xffff0000u);
            const u32 hp = pack_hi16(u1, u0);
            const u32 lp = pack_hi16(__float_as_uint(x1 - hf1),
                                     __float_as_uint(x0 - hf0));
            *(u32*)&Xh[sn * KSTRIDE + k] = hp;              // 4B-aligned b32 stores
            *(u32*)&Xl[sn * KSTRIDE + k] = lp;
        }
        // ---- prefetch next slab's rows while this one computes ----
        if (k0 + KCHUNK < KB) {
            const int kn = k0 + KCHUNK;
            const float* xn = (kn < 128) ? (e_vw + xbase + (size_t)kn * NB)
                                         : (h_w  + xbase + (size_t)(kn - 128) * NB);
#pragma unroll
            for (int j = 0; j < 8; ++j) {
                const int k = (kg * 8 + j) * 2;
                __builtin_prefetch(xn + (size_t)k * NB + n0 + sn, 0, 0);
                __builtin_prefetch(xn + (size_t)(k + 1) * NB + n0 + sn, 0, 0);
            }
        }
        __syncthreads();

        // ---- A fragments: 2 strips x {hi,lo} planes from pre-split weights ----
        // 16-bit A 16x32 layout: lane row = strip+lane%16; K = {8*hi..+7, +16..+23}
        v16bf ah[2], al[2];
#pragma unroll
        for (int s = 0; s < 2; ++s) {
            const int woff = ((m0 + s * 16 + lane_lo) << 8) + k0 + (hi << 3);
            ah[s] = join16(*(const v8us*)(Wh + woff), *(const v8us*)(Wh + woff + 16));
            al[s] = join16(*(const v8us*)(Wl + woff), *(const v8us*)(Wl + woff + 16));
        }

        // ---- per n-subtile: B fragments (2 x ds_load_b128 per plane) feed 6 WMMAs ----
        // 16-bit B 32x16 layout: lane col = nw+nt*16+lane%16; K = {8*hi..+7, +16..+23}
#pragma unroll
        for (int nt = 0; nt < 4; ++nt) {
            const int base = (nw + nt * 16 + lane_lo) * KSTRIDE + 8 * hi;
            const v16bf bh = join16(*(const v8us*)&Xh[base], *(const v8us*)&Xh[base + 16]);
            const v16bf bl = join16(*(const v8us*)&Xl[base], *(const v8us*)&Xl[base + 16]);
#pragma unroll
            for (int s = 0; s < 2; ++s) {
                acc[s][nt] = __builtin_amdgcn_wmma_f32_16x16x32_bf16(
                    false, ah[s], false, bh, (short)0, acc[s][nt], false, false);
                acc[s][nt] = __builtin_amdgcn_wmma_f32_16x16x32_bf16(
                    false, ah[s], false, bl, (short)0, acc[s][nt], false, false);
                acc[s][nt] = __builtin_amdgcn_wmma_f32_16x16x32_bf16(
                    false, al[s], false, bh, (short)0, acc[s][nt], false, false);
            }
        }
    }

    // ---- epilogue: each row-store covers two 64B segments ----
    float* obase = out + (size_t)b * (size_t)(MB * NB) + n0 + nw;
#pragma unroll
    for (int s = 0; s < 2; ++s)
#pragma unroll
        for (int nt = 0; nt < 4; ++nt)
#pragma unroll
            for (int i = 0; i < 8; ++i) {
                const int m = m0 + s * 16 + i + 8 * hi;
                obase[(size_t)m * NB + nt * 16 + lane_lo] = acc[s][nt][i];
            }
}

extern "C" void kernel_launch(void* const* d_in, const int* in_sizes, int n_in,
                              void* d_out, int out_size, void* d_ws, size_t ws_size,
                              hipStream_t stream) {
    // setup_inputs order: h_v(unused), h_w, e_vw, W_e, b_e, W_h, b_h  (all fp32)
    const float* h_w  = (const float*)d_in[1];
    const float* e_vw = (const float*)d_in[2];
    const float* W_e  = (const float*)d_in[3];
    const float* b_e  = (const float*)d_in[4];
    const float* W_h  = (const float*)d_in[5];
    const float* b_h  = (const float*)d_in[6];
    float* out = (float*)d_out;

    // d_ws layout: Wh bf16 [128][256] (64KB) | Wl bf16 [128][256] (64KB)
    u16* Wh = (u16*)d_ws;
    u16* Wl = Wh + MB * KB;

    hipLaunchKernelGGL(split_weights, dim3(MB * KB / 256), dim3(256), 0, stream,
                       W_e, W_h, Wh, Wl);
    hipLaunchKernelGGL(msg_gemm_bf16x3, dim3(NB / NTILE, 1024), dim3(256), 0, stream,
                       h_w, e_vw, Wh, Wl, b_e, b_h, out);
}
